// TourInterpretableGraphACTNN_82935818486082
// MI455X (gfx1250) — matrix operand order repaired and checked by reading, not compiled, not run on hardware
//
#include <hip/hip_runtime.h>
#include <hip/hip_bf16.h>

// Dims from the reference
#define Zz    8000
#define Bb    4096
#define FZz   64
#define Pp    10
#define Mm    8
#define Ss    16
#define Hh    128
#define ALPHA 0.7f
#define NEGV  -1e9f

#define FT_STRIDE 8008   // padded ftT row length (multiple of 8)

typedef __attribute__((ext_vector_type(16))) __bf16        v16bf;
typedef __attribute__((ext_vector_type(8)))  float         v8f;
typedef __attribute__((ext_vector_type(4)))  float         f32x4;
typedef __attribute__((ext_vector_type(4)))  unsigned int  u32x4;
typedef __attribute__((ext_vector_type(2)))  unsigned int  u32x2;
typedef __attribute__((ext_vector_type(8)))  int           i32x8;
typedef __attribute__((ext_vector_type(4)))  int           i32x4;

union Frag { v16bf v; u32x4 q[2]; };
union Pk4  { __bf16 h[4]; u32x2 u; };

// ---------------- prep kernel: support bitmask + transposed bf16 features ----
__global__ __launch_bounds__(256) void prep_kernel(
    const float* __restrict__ zpm, const float* __restrict__ zfm,
    unsigned short* __restrict__ mask16, __bf16* __restrict__ ftT)
{
  const int n1 = Zz;
  const int n2 = FZz * FT_STRIDE;
  for (int i = blockIdx.x * blockDim.x + threadIdx.x; i < n1 + n2;
       i += gridDim.x * blockDim.x) {
    if (i < n1) {
      unsigned v = 0;
      #pragma unroll
      for (int p = 0; p < Pp; ++p) v |= (zpm[i * Pp + p] > 0.0f ? 1u : 0u) << p;
      mask16[i] = (unsigned short)v;
    } else {
      int e = i - n1;
      int c = e / FT_STRIDE, z = e % FT_STRIDE;
      ftT[e] = (__bf16)((z < Zz) ? zfm[(size_t)z * FZz + c] : 0.0f);
    }
  }
}

// ---------------- fused main kernel ----------------
template <bool WSP>
__global__ __launch_bounds__(256) void tour_fused_kernel(
    const float* __restrict__ zfm, const float* __restrict__ zpm,
    const float* __restrict__ od,  const float* __restrict__ psp,
    const float* __restrict__ msp, const float* __restrict__ pmp,
    const float* __restrict__ olm, const float* __restrict__ relw,
    const float* __restrict__ W1,  const float* __restrict__ b1,
    const float* __restrict__ Wp,  const float* __restrict__ Wm,
    const int* __restrict__ org_g, const int* __restrict__ pid_g,
    const int* __restrict__ step_g, const int* __restrict__ rid_g,
    const unsigned short* __restrict__ mask16, const __bf16* __restrict__ ftT,
    float* __restrict__ out)
{
  __shared__ int   s_org[16], s_pid[16], s_step[16], s_rid[16];
  __shared__ float s_m[16], s_sinv[16];
  __shared__ __align__(16) float  tctx[16 * 64];
  __shared__ __align__(16) float  gatel[16 * 64];
  __shared__ __align__(16) __bf16 w1t[128 * 72];
  __shared__ __align__(16) __bf16 p_lds[16 * 136];
  __shared__ __align__(16) __bf16 ft_lds[64 * 136];
  __shared__ __align__(16) __bf16 x_lds[16 * 72];
  __shared__ __align__(16) float  h_lds[16 * 128];
  __shared__ __align__(16) float  red[4 * 256];

  const int tid  = threadIdx.x;
  const int lane = tid & 31;
  const int w    = tid >> 5;
  const int tile = blockIdx.x;

  if (tid < 16) {
    int b = tile * 16 + tid;
    s_org[tid]  = org_g[b];
    s_pid[tid]  = pid_g[b];
    s_step[tid] = step_g[b];
    s_rid[tid]  = rid_g[b];
  }
  __syncthreads();

  // ---- Phase 0: gate + W1^T staging ----
  #pragma unroll
  for (int i = 0; i < 4; ++i) {
    int e = tid + i * 256; int r = e >> 6, c = e & 63;
    float g = relw[s_rid[r] * FZz + c];
    gatel[e] = 1.0f / (1.0f + __expf(-g));
  }
  #pragma unroll
  for (int i = 0; i < 32; ++i) {
    int e = tid + i * 256; int k = e >> 7, c = e & 127;
    w1t[c * 72 + k] = (__bf16)W1[e];
  }

  // ---- Phase 1: softmax stats + top-8 (1 wave : 2 tours), float4-vectorized
  for (int tt = 0; tt < 2; ++tt) {
    const int t  = w * 2 + tt;
    const int oz = s_org[t], pp = s_pid[t];
    const float* odr = od + (size_t)oz * Zz;
    float m = -INFINITY, s = 0.0f;
    float tv[8]; int ti8[8];
    #pragma unroll
    for (int j = 0; j < 8; ++j) { tv[j] = -INFINITY; ti8[j] = 0; }

    for (int it = 0; it <= 62; ++it) {
      int z;
      if (it < 62) z = it * 128 + lane * 4;
      else { if (lane >= 16) break; z = 62 * 128 + lane * 4; }   // tail 64
      f32x4 o = *(const f32x4*)(odr + z);
      f32x4 l = *(const f32x4*)(olm + z);
      float xs[4];
      if (WSP) {
        u32x2 mw = *(const u32x2*)(mask16 + z);
        unsigned mm[4] = { mw.x & 0xFFFFu, mw.x >> 16, mw.y & 0xFFFFu, mw.y >> 16 };
        #pragma unroll
        for (int j = 0; j < 4; ++j)
          xs[j] = o[j] + l[j] + (((mm[j] >> pp) & 1u) ? 0.0f : NEGV);
      } else {
        #pragma unroll
        for (int j = 0; j < 4; ++j)
          xs[j] = o[j] + l[j] + (zpm[(z + j) * Pp + pp] > 0.0f ? 0.0f : NEGV);
      }
      #pragma unroll
      for (int j = 0; j < 4; ++j) {
        float x = xs[j];
        float mn = fmaxf(m, x);
        s = s * __expf(m - mn) + __expf(x - mn);
        m = mn;
        if (x > tv[7]) {
          tv[7] = x; ti8[7] = z + j;
          #pragma unroll
          for (int q = 7; q > 0; --q)
            if (tv[q] > tv[q - 1]) {
              float a = tv[q]; tv[q] = tv[q - 1]; tv[q - 1] = a;
              int   b = ti8[q]; ti8[q] = ti8[q - 1]; ti8[q - 1] = b;
            }
        }
      }
    }
    #pragma unroll
    for (int off = 16; off >= 1; off >>= 1) {
      float m2 = __shfl_xor(m, off, 32);
      float s2 = __shfl_xor(s, off, 32);
      float mn = fmaxf(m, m2);
      s = s * __expf(m - mn) + s2 * __expf(m2 - mn);
      m = mn;
    }
    float wv8[8]; int wi8[8];
    #pragma unroll
    for (int r8 = 0; r8 < 8; ++r8) {
      float cv = tv[0]; int ci = ti8[0]; int cl = lane;
      #pragma unroll
      for (int off = 16; off >= 1; off >>= 1) {
        float ov = __shfl_xor(cv, off, 32);
        int   oi = __shfl_xor(ci, off, 32);
        int   ol = __shfl_xor(cl, off, 32);
        if (ov > cv || (ov == cv && ol < cl)) { cv = ov; ci = oi; cl = ol; }
      }
      wv8[r8] = cv; wi8[r8] = ci;
      if (cl == lane) {
        #pragma unroll
        for (int j = 0; j < 7; ++j) { tv[j] = tv[j + 1]; ti8[j] = ti8[j + 1]; }
        tv[7] = -INFINITY;
      }
    }
    float se = 0.0f, wk[8];
    #pragma unroll
    for (int j = 0; j < 8; ++j) { wk[j] = __expf(wv8[j] - wv8[0]); se += wk[j]; }
    float inv = 1.0f / se;
    float a0 = 0.0f, a1 = 0.0f;
    #pragma unroll
    for (int j = 0; j < 8; ++j) {
      const float* fr = zfm + (size_t)wi8[j] * FZz;
      float wj = wk[j] * inv;
      a0 += wj * fr[lane];
      a1 += wj * fr[lane + 32];
    }
    tctx[t * 64 + lane]      = ALPHA * a0;
    tctx[t * 64 + lane + 32] = ALPHA * a1;
    if (lane == 0) { s_m[t] = m; s_sinv[t] = 1.0f / s; }
  }
  __syncthreads();

  // ---- Phase 2: full-context GEMM over Z, bf16 WMMA, TDM-fed B tiles ----
  const int wt  = w & 3;
  const int kh  = w >> 2;
  const int l16 = lane & 15;
  const int hi  = lane >> 4;
  v8f acc = {};

  for (int zb = 0; zb < Zz; zb += 128) {
    if (WSP) {
#if __has_builtin(__builtin_amdgcn_tensor_load_to_lds) && __has_builtin(__builtin_amdgcn_s_wait_tensorcnt)
      if (w == 0) {
        // Tensor DMA: ftT[64][FT_STRIDE] bf16 tile (64 x 128) -> ft_lds,
        // LDS padded by 4 DWORDs every 64 DWORDs => row stride 136 elements.
        unsigned ldsoff = (unsigned)(unsigned long long)(&ft_lds[0]);
        unsigned long long ga =
            (unsigned long long)(ftT) + (unsigned long long)zb * 2ull;
        unsigned td0 = (unsigned)(FT_STRIDE - zb);   // remaining row length
        u32x4 g0 = { 1u,                              // count=1 descriptor
                     ldsoff,                          // lds_addr
                     (unsigned)(ga & 0xFFFFFFFFu),
                     (unsigned)((ga >> 32) & 0x01FFFFFFu) | (2u << 30) }; // type=2
        i32x8 g1;
        g1[0] = (int)((1u << 16) |        // data_size = 2B
                      (1u << 20) |        // pad_enable
                      (5u << 22) |        // pad_interval: 64 DWORDs
                      (3u << 25));        // pad_amount: 4 DWORDs
        g1[1] = (int)((td0 & 0xFFFFu) << 16);                 // tensor_dim0 lo
        g1[2] = (int)(((td0 >> 16) & 0xFFFFu) | (64u << 16)); // dim0 hi | tensor_dim1=64
        g1[3] = (int)(128u << 16);                            // tile_dim0 = 128
        g1[4] = (int)(64u);                                   // tile_dim1 = 64
        g1[5] = (int)FT_STRIDE;                               // tensor_dim0_stride lo
        g1[6] = 0; g1[7] = 0;
        i32x4 gz4 = { 0, 0, 0, 0 };
        i32x8 gz8 = { 0, 0, 0, 0, 0, 0, 0, 0 };
        __builtin_amdgcn_tensor_load_to_lds(g0, g1, gz4, gz4, gz8, 0);
      }
#else
      // No TDM builtin: plain vectorized copy of precomputed bf16 tiles.
      #pragma unroll
      for (int i = 0; i < 4; ++i) {
        int q = tid + i * 256;            // 1024 x 8 elements
        int r = q >> 4, k8 = (q & 15) * 8;
        *(u32x4*)(ft_lds + r * 136 + k8) =
            *(const u32x4*)(ftT + (size_t)r * FT_STRIDE + zb + k8);
      }
#endif
    } else {
      #pragma unroll
      for (int i = 0; i < 32; ++i) {
        int e = tid + i * 256; int dz = e >> 6, c = e & 63;
        int z = zb + dz;
        float fv = (z < Zz) ? zfm[(size_t)z * FZz + c] : 0.0f;
        ft_lds[c * 136 + dz] = (__bf16)fv;
      }
    }

    // probs chunk (A matrix), float4-vectorized: 16 tours x 128 k
    #pragma unroll
    for (int i = 0; i < 2; ++i) {
      int q = tid + i * 256;              // 0..511
      int r = q >> 5;                     // tour
      int k4 = (q & 31) * 4;              // 0..124
      int z = zb + k4;                    // Z % 4 == 0 -> whole-vec bounds test
      Pk4 pk;
      if (z < Zz) {
        f32x4 o = *(const f32x4*)(od + (size_t)s_org[r] * Zz + z);
        f32x4 l = *(const f32x4*)(olm + z);
        float mr = s_m[r], sr = s_sinv[r];
        int pp = s_pid[r];
        if (WSP) {
          u32x2 mw = *(const u32x2*)(mask16 + z);
          unsigned mm[4] = { mw.x & 0xFFFFu, mw.x >> 16, mw.y & 0xFFFFu, mw.y >> 16 };
          #pragma unroll
          for (int j = 0; j < 4; ++j) {
            float x = o[j] + l[j] + (((mm[j] >> pp) & 1u) ? 0.0f : NEGV);
            pk.h[j] = (__bf16)(__expf(x - mr) * sr);
          }
        } else {
          #pragma unroll
          for (int j = 0; j < 4; ++j) {
            float x = o[j] + l[j] + (zpm[(z + j) * Pp + pp] > 0.0f ? 0.0f : NEGV);
            pk.h[j] = (__bf16)(__expf(x - mr) * sr);
          }
        }
      } else {
        pk.u = (u32x2){0u, 0u};
      }
      *(u32x2*)(p_lds + r * 136 + k4) = pk.u;
    }

#if __has_builtin(__builtin_amdgcn_tensor_load_to_lds) && __has_builtin(__builtin_amdgcn_s_wait_tensorcnt)
    if (WSP && w == 0) __builtin_amdgcn_s_wait_tensorcnt(0);
#endif
    __syncthreads();

    #pragma unroll
    for (int ks = 0; ks < 2; ++ks) {
      int kb = (kh * 2 + ks) * 32;
      Frag a, b;
      const __bf16* pr = &p_lds[l16 * 136 + kb + hi * 8];
      a.q[0] = *(const u32x4*)pr;
      a.q[1] = *(const u32x4*)(pr + 16);
      const __bf16* br = &ft_lds[(wt * 16 + l16) * 136 + kb + hi * 8];
      b.q[0] = *(const u32x4*)br;
      b.q[1] = *(const u32x4*)(br + 16);
      acc = __builtin_amdgcn_wmma_f32_16x16x32_bf16(
          false, a.v, false, b.v, (short)0, acc, false, false);
    }
    __syncthreads();
  }

  // ---- K-half reduction + blend + gate -> x (bf16) ----
  if (w >= 4) {
    #pragma unroll
    for (int j = 0; j < 8; ++j) red[(w - 4) * 256 + j * 32 + lane] = acc[j];
  }
  __syncthreads();
  if (w < 4) {
    #pragma unroll
    for (int j = 0; j < 8; ++j) {
      float full = acc[j] + red[w * 256 + j * 32 + lane];
      int row = j + hi * 8;
      int col = wt * 16 + l16;
      float ctx = tctx[row * 64 + col] + (1.0f - ALPHA) * full;
      x_lds[row * 72 + col] = (__bf16)(ctx * gatel[row * 64 + col]);
    }
  }
  __syncthreads();

  // ---- Hidden layer: [16,64] @ [64,128] via bf16 WMMA ----
  v8f hacc = {};
  #pragma unroll
  for (int ks = 0; ks < 2; ++ks) {
    int kb = ks * 32;
    Frag a, bf;
    const __bf16* pr = &x_lds[l16 * 72 + kb + hi * 8];
    a.q[0] = *(const u32x4*)pr;
    a.q[1] = *(const u32x4*)(pr + 16);
    const __bf16* br = &w1t[(w * 16 + l16) * 72 + kb + hi * 8];
    bf.q[0] = *(const u32x4*)br;
    bf.q[1] = *(const u32x4*)(br + 16);
    hacc = __builtin_amdgcn_wmma_f32_16x16x32_bf16(
        false, a.v, false, bf.v, (short)0, hacc, false, false);
  }
  #pragma unroll
  for (int j = 0; j < 8; ++j) {
    int row = j + hi * 8;
    int col = w * 16 + l16;
    h_lds[row * 128 + col] = fmaxf(hacc[j] + b1[col], 0.0f);
  }
  __syncthreads();

  // ---- Heads ----
  for (int o = tid; o < 16 * 18; o += 256) {
    int r = o / 18, c = o % 18;
    const float* hr = &h_lds[r * 128];
    float dot = 0.0f;
    if (c < Pp) {
      #pragma unroll 4
      for (int k = 0; k < Hh; ++k) dot += hr[k] * Wp[k * Pp + c];
      dot += psp[c * Ss + s_step[r]];
    } else {
      int cm = c - Pp;
      #pragma unroll 4
      for (int k = 0; k < Hh; ++k) dot += hr[k] * Wm[k * Mm + cm];
      dot += pmp[s_pid[r] * Mm + cm] + msp[cm * Ss + s_step[r]];
    }
    out[((size_t)tile * 16 + r) * 18 + c] = dot;
  }
}

extern "C" void kernel_launch(void* const* d_in, const int* in_sizes, int n_in,
                              void* d_out, int out_size, void* d_ws, size_t ws_size,
                              hipStream_t stream) {
  const float* zfm  = (const float*)d_in[0];
  const float* zpm  = (const float*)d_in[1];
  const float* od   = (const float*)d_in[2];
  const float* psp  = (const float*)d_in[3];
  const float* msp  = (const float*)d_in[4];
  const float* pmp  = (const float*)d_in[5];
  const float* olm  = (const float*)d_in[6];
  const float* relw = (const float*)d_in[7];
  const float* W1   = (const float*)d_in[8];
  const float* b1   = (const float*)d_in[9];
  const float* Wp   = (const float*)d_in[10];
  const float* Wm   = (const float*)d_in[11];
  const int*   org  = (const int*)d_in[12];
  const int*   pid  = (const int*)d_in[13];
  const int*   stp  = (const int*)d_in[14];
  const int*   rid  = (const int*)d_in[15];
  float*       out  = (float*)d_out;

  const size_t maskBytes = 16384;                        // 8000 x u16, padded
  const size_t ftBytes   = (size_t)FZz * FT_STRIDE * 2;  // bf16
  const bool   wsp       = (ws_size >= maskBytes + ftBytes);

  dim3 grid(Bb / 16), block(256);
  if (wsp) {
    unsigned short* mask16 = (unsigned short*)d_ws;
    __bf16* ftT = (__bf16*)((char*)d_ws + maskBytes);
    prep_kernel<<<dim3(1024), dim3(256), 0, stream>>>(zpm, zfm, mask16, ftT);
    tour_fused_kernel<true><<<grid, block, 0, stream>>>(
        zfm, zpm, od, psp, msp, pmp, olm, relw, W1, b1, Wp, Wm,
        org, pid, stp, rid, mask16, ftT, out);
  } else {
    tour_fused_kernel<false><<<grid, block, 0, stream>>>(
        zfm, zpm, od, psp, msp, pmp, olm, relw, W1, b1, Wp, Wm,
        org, pid, stp, rid, nullptr, nullptr, out);
  }
}
